// TransformerConv_9311489097786
// MI455X (gfx1250) — matrix operand order repaired and checked by reading, not compiled, not run on hardware
//
#include <hip/hip_runtime.h>
#include <hip/hip_bf16.h>

// ---------------------------------------------------------------------------
// Problem constants (match reference)
// ---------------------------------------------------------------------------
#define NN      50000
#define EE      800000
#define DIM     64
#define NLAYERS 4
#define NG      64
#define NOUT    10

typedef __attribute__((ext_vector_type(16))) __bf16 v16bf;
typedef __attribute__((ext_vector_type(8)))  __bf16 v8bf;
typedef __attribute__((ext_vector_type(8)))  float  v8f;

// ---------------------------------------------------------------------------
// fp32 -> bf16 conversion of the hidden state (A-matrix feed for WMMA)
// ---------------------------------------------------------------------------
__global__ __launch_bounds__(256) void convert_bf16_k(const float* __restrict__ in,
                                                      __bf16* __restrict__ out, int n8) {
    int t = blockIdx.x * blockDim.x + threadIdx.x;
    if (t >= n8) return;
    const float* p = in + (size_t)t * 8;
    v8bf o;
#pragma unroll
    for (int i = 0; i < 8; ++i) o[i] = (__bf16)p[i];
    *(v8bf*)(out + (size_t)t * 8) = o;
}

// ---------------------------------------------------------------------------
// Fused q/k/v/s projection: one wave per 16-row tile, WMMA bf16->f32.
// Weights staged transposed (Wt[n][k]) as bf16 in LDS; bias fused.
// s-projection (+bs) is written to hnext so edge_agg can atomically add onto it.
// ---------------------------------------------------------------------------
__global__ __launch_bounds__(256) void gemm_proj_k(
    const __bf16* __restrict__ hbf,
    const float* __restrict__ Wq, const float* __restrict__ Wk,
    const float* __restrict__ Wv, const float* __restrict__ Ws,
    const float* __restrict__ bq, const float* __restrict__ bk,
    const float* __restrict__ bv, const float* __restrict__ bs,
    float* __restrict__ q, float* __restrict__ k,
    float* __restrict__ v, float* __restrict__ hnext, int nTiles)
{
    __shared__ __align__(32) __bf16 Wt[4][DIM][DIM];   // 32 KB, transposed
    __shared__ float bias[4][DIM];

    const float* Wsrc[4] = {Wq, Wk, Wv, Ws};
    const float* bsrc[4] = {bq, bk, bv, bs};
    const int tid = threadIdx.x;

#pragma unroll
    for (int p = 0; p < 4; ++p) {
        const float* W = Wsrc[p];
        for (int i = tid; i < DIM * DIM; i += 256) {
            int kk = i >> 6, n = i & 63;          // W is [k][n] row-major
            Wt[p][n][kk] = (__bf16)W[i];
        }
    }
    { int p = tid >> 6, n = tid & 63; bias[p][n] = bsrc[p][n]; }
    __syncthreads();

    const int wave = tid >> 5, lane = tid & 31;
    const int mtile = blockIdx.x * 8 + wave;
    if (mtile >= nTiles) return;                   // wave-uniform: EXEC stays all-ones

    const int baseRow = mtile << 4;
    const int mlane   = lane & 15;
    const int hiHalf  = lane >> 4;                 // 0 or 1
    const int koff    = hiHalf << 4;               // lane-half K offset in A layout
    const __bf16* rowp = hbf + (size_t)(baseRow + mlane) * DIM;

    // A fragments for k-steps 0 (K=0..31) and 1 (K=32..63)
    v16bf a[2];
#pragma unroll
    for (int ks = 0; ks < 2; ++ks) {
        v8bf lo = *(const v8bf*)(rowp + 32 * ks + koff);
        v8bf hi = *(const v8bf*)(rowp + 32 * ks + 8 + koff);
        a[ks] = __builtin_shufflevector(lo, hi, 0,1,2,3,4,5,6,7,8,9,10,11,12,13,14,15);
    }

    const int kbase = hiHalf << 4;                 // B layout: hi lanes take K+16
#pragma unroll
    for (int p = 0; p < 4; ++p) {
        float* outp = (p == 0) ? q : (p == 1) ? k : (p == 2) ? v : hnext;
#pragma unroll
        for (int nt = 0; nt < 4; ++nt) {
            const int n = (nt << 4) + mlane;
            v16bf b0 = *(const v16bf*)(&Wt[p][n][kbase]);
            v16bf b1 = *(const v16bf*)(&Wt[p][n][32 + kbase]);
            v8f c = {};
            c = __builtin_amdgcn_wmma_f32_16x16x32_bf16(false, a[0], false, b0,
                                                        (short)0, c, false, false);
            c = __builtin_amdgcn_wmma_f32_16x16x32_bf16(false, a[1], false, b1,
                                                        (short)0, c, false, false);
            const float bval = bias[p][n];
            const int m0 = baseRow + (hiHalf << 3); // C/D layout: M = r + 8*(lane/16)
#pragma unroll
            for (int r = 0; r < 8; ++r)
                outp[(size_t)(m0 + r) * DIM + n] = c[r] + bval;
        }
    }
}

// ---------------------------------------------------------------------------
// Per-layer segment softmax + aggregation
// ---------------------------------------------------------------------------
__global__ __launch_bounds__(256) void init_nodes_k(float* __restrict__ m,
                                                    float* __restrict__ z, int n) {
    int t = blockIdx.x * blockDim.x + threadIdx.x;
    if (t < n) { m[t] = -__builtin_inff(); z[t] = 0.f; }
}

__device__ __forceinline__ void atomicMaxF32(float* addr, float val) {
    if (val >= 0.f) atomicMax((int*)addr, __float_as_int(val));
    else            atomicMin((unsigned int*)addr, (unsigned int)__float_as_int(val));
}

__global__ __launch_bounds__(256) void edge_alpha_k(
    const int* __restrict__ src, const int* __restrict__ dst,
    const float* __restrict__ q, const float* __restrict__ k,
    float* __restrict__ alpha, float* __restrict__ mmax, int nE)
{
    int e = blockIdx.x * blockDim.x + threadIdx.x;
    if (e >= nE) return;
    int dN = dst[e], sN = src[e];
    const float4* qr = (const float4*)(q + (size_t)dN * DIM);
    const float4* kr = (const float4*)(k + (size_t)sN * DIM);
    float acc = 0.f;
#pragma unroll
    for (int j = 0; j < 16; ++j) {
        float4 a = qr[j], b = kr[j];
        acc += a.x * b.x + a.y * b.y + a.z * b.z + a.w * b.w;
    }
    acc *= 0.125f;   // 1/sqrt(64)
    alpha[e] = acc;
    atomicMaxF32(&mmax[dN], acc);
}

__global__ __launch_bounds__(256) void edge_exp_k(
    const int* __restrict__ dst, float* __restrict__ alpha,
    const float* __restrict__ mmax, float* __restrict__ z, int nE)
{
    int e = blockIdx.x * blockDim.x + threadIdx.x;
    if (e >= nE) return;
    int d = dst[e];
    float ex = __expf(alpha[e] - mmax[d]);
    alpha[e] = ex;
    atomicAdd(&z[d], ex);
}

__global__ __launch_bounds__(256) void edge_agg_k(
    const int* __restrict__ src, const int* __restrict__ dst,
    const float* __restrict__ alpha, const float* __restrict__ z,
    const float* __restrict__ v, float* __restrict__ hnext, int nE)
{
    int t = blockIdx.x * blockDim.x + threadIdx.x;
    int e = t >> 4;
    if (e >= nE) return;
    int fg = (t & 15) << 2;
    int sN = src[e], dN = dst[e];
    float w = alpha[e] / (z[dN] + 1e-16f);
    float4 vv = *(const float4*)(v + (size_t)sN * DIM + fg);
    float* dp = hnext + (size_t)dN * DIM + fg;
    atomicAdd(dp + 0, w * vv.x);
    atomicAdd(dp + 1, w * vv.y);
    atomicAdd(dp + 2, w * vv.z);
    atomicAdd(dp + 3, w * vv.w);
}

// ---------------------------------------------------------------------------
// Pooling + final linear
// ---------------------------------------------------------------------------
__global__ __launch_bounds__(256) void zero_f32_k(float* __restrict__ p, int n) {
    int t = blockIdx.x * blockDim.x + threadIdx.x;
    if (t < n) p[t] = 0.f;
}

__global__ __launch_bounds__(256) void pool_k(
    const float* __restrict__ h, const int* __restrict__ batch,
    float* __restrict__ pooled, float* __restrict__ cnt, int nN)
{
    int t = blockIdx.x * blockDim.x + threadIdx.x;
    int n = t >> 4;
    if (n >= nN) return;
    int fg = (t & 15) << 2;
    int g = batch[n];
    float4 hv = *(const float4*)(h + (size_t)n * DIM + fg);
    float* pp = pooled + (size_t)g * DIM + fg;
    atomicAdd(pp + 0, hv.x);
    atomicAdd(pp + 1, hv.y);
    atomicAdd(pp + 2, hv.z);
    atomicAdd(pp + 3, hv.w);
    if ((t & 15) == 0) atomicAdd(&cnt[g], 1.0f);
}

__global__ __launch_bounds__(640) void final_k(
    const float* __restrict__ pooled, const float* __restrict__ cnt,
    const float* __restrict__ Wf, const float* __restrict__ bf_,
    float* __restrict__ out)
{
    int t = threadIdx.x;            // 0 .. 639
    int g = t / NOUT, o = t % NOUT;
    float inv = 1.f / fmaxf(cnt[g], 1.f);
    float acc = bf_[o];
#pragma unroll 8
    for (int kk = 0; kk < DIM; ++kk)
        acc += pooled[g * DIM + kk] * inv * Wf[kk * NOUT + o];
    out[t] = acc;
}

// ---------------------------------------------------------------------------
// Host-side orchestration
// ---------------------------------------------------------------------------
static inline size_t alignUp(size_t x) { return (x + 255) & ~(size_t)255; }

extern "C" void kernel_launch(void* const* d_in, const int* in_sizes, int n_in,
                              void* d_out, int out_size, void* d_ws, size_t ws_size,
                              hipStream_t stream) {
    const float* x     = (const float*)d_in[0];
    const int*   ei    = (const int*)d_in[1];     // [2][EE]
    const int*   batch = (const int*)d_in[2];
    const float* Wq    = (const float*)d_in[3];
    const float* bq    = (const float*)d_in[4];
    const float* Wk    = (const float*)d_in[5];
    const float* bk    = (const float*)d_in[6];
    const float* Wv    = (const float*)d_in[7];
    const float* bv    = (const float*)d_in[8];
    const float* Ws    = (const float*)d_in[9];
    const float* bs    = (const float*)d_in[10];
    const float* Wf    = (const float*)d_in[11];
    const float* bf_   = (const float*)d_in[12];
    float* out = (float*)d_out;

    const int* src = ei;
    const int* dst = ei + EE;

    // workspace carve-up
    char* ws = (char*)d_ws;
    size_t off = 0;
    auto take = [&](size_t bytes) { char* p = ws + off; off += alignUp(bytes); return p; };
    float*  hA     = (float*) take((size_t)NN * DIM * 4);
    float*  hB     = (float*) take((size_t)NN * DIM * 4);
    __bf16* hbf    = (__bf16*)take((size_t)NN * DIM * 2);
    float*  qb     = (float*) take((size_t)NN * DIM * 4);
    float*  kb     = (float*) take((size_t)NN * DIM * 4);
    float*  vb     = (float*) take((size_t)NN * DIM * 4);
    float*  alphab = (float*) take((size_t)EE * 4);
    float*  mmax   = (float*) take((size_t)NN * 4);
    float*  zsum   = (float*) take((size_t)NN * 4);
    float*  pooled = (float*) take((size_t)(NG * DIM + NG) * 4); // pooled + cnt contiguous
    float*  cnt    = pooled + NG * DIM;
    if (off > ws_size) return;  // workspace too small: bail deterministically

    const int n8        = NN * DIM / 8;          // 400000
    const int nTiles    = NN / 16;               // 3125
    const int gConv     = (n8 + 255) / 256;
    const int gGemm     = (nTiles + 7) / 8;
    const int gNodes    = (NN + 255) / 256;
    const int gEdges    = (EE + 255) / 256;
    const int gEdges16  = ((EE * 16) + 255) / 256;
    const int gNodes16  = ((NN * 16) + 255) / 256;

    const float* cur = x;
    float* nxt = hA;
    float* other = hB;

    for (int l = 0; l < NLAYERS; ++l) {
        const float* wq = Wq + (size_t)l * DIM * DIM;
        const float* wk = Wk + (size_t)l * DIM * DIM;
        const float* wv = Wv + (size_t)l * DIM * DIM;
        const float* wss = Ws + (size_t)l * DIM * DIM;
        const float* bql = bq + (size_t)l * DIM;
        const float* bkl = bk + (size_t)l * DIM;
        const float* bvl = bv + (size_t)l * DIM;
        const float* bsl = bs + (size_t)l * DIM;

        convert_bf16_k<<<gConv, 256, 0, stream>>>(cur, hbf, n8);
        gemm_proj_k<<<gGemm, 256, 0, stream>>>(hbf, wq, wk, wv, wss,
                                               bql, bkl, bvl, bsl,
                                               qb, kb, vb, nxt, nTiles);
        init_nodes_k<<<gNodes, 256, 0, stream>>>(mmax, zsum, NN);
        edge_alpha_k<<<gEdges, 256, 0, stream>>>(src, dst, qb, kb, alphab, mmax, EE);
        edge_exp_k<<<gEdges, 256, 0, stream>>>(dst, alphab, mmax, zsum, EE);
        edge_agg_k<<<gEdges16, 256, 0, stream>>>(src, dst, alphab, zsum, vb, nxt, EE);

        cur = nxt;
        nxt = other;
        other = (float*)cur;   // ping-pong (layer 0 leaves x untouched)
        if (l == 0) { nxt = hB; other = hA; }
    }

    zero_f32_k<<<(NG * DIM + NG + 255) / 256, 256, 0, stream>>>(pooled, NG * DIM + NG);
    pool_k<<<gNodes16, 256, 0, stream>>>(cur, batch, pooled, cnt, NN);
    final_k<<<1, NG * NOUT, 0, stream>>>(pooled, cnt, Wf, bf_, out);
    (void)in_sizes; (void)n_in; (void)out_size;
}
